// VanilaRNN_39213051412509
// MI455X (gfx1250) — matrix operand order repaired and checked by reading, not compiled, not run on hardware
//
#include <hip/hip_runtime.h>
#include <stdint.h>
#include <stddef.h>

typedef __attribute__((ext_vector_type(16))) __bf16 v16bf;
typedef __attribute__((ext_vector_type(8)))  float  v8f;

// Explicit global-address-space access (forces global_load/global_store, frees
// the flat/LDS path and enables saddr+imm addressing).
#define GCAST(T, p) ((__attribute__((address_space(1))) T*)(p))

namespace cfg {
constexpr int B = 64, T = 256, DIN = 512, H = 1024, DOUT = 512;
constexpr int KT0  = (DIN + H) / 32;  // 48 k-tiles, layer0
constexpr int KT1  = (H + H) / 32;    // 64 k-tiles, layer1
constexpr int KTFC = H / 32;          // 32 k-tiles, fc
constexpr int NT01 = (4 * H) / 16;    // 256 n-tiles (4 gates x 64)
constexpr int NTFC = DOUT / 16;       // 32 n-tiles
constexpr int NBLK = 64;              // persistent grid
constexpr int TPB  = 256;             // 8 wave32 per block
constexpr float SCALE0 = 0.02551551815399144f;  // 1/sqrt(1536)
constexpr float SCALE1 = 0.02209708691207961f;  // 1/sqrt(2048)
constexpr size_t A0E = (size_t)4 * KT0 * 512;   // u16 elems per parity
constexpr size_t A1E = (size_t)4 * KT1 * 512;
}

__device__ __forceinline__ v8f wmma_bf16(v16bf a, v16bf b, v8f c) {
  return __builtin_amdgcn_wmma_f32_16x16x32_bf16(
      /*neg_a=*/false, a, /*neg_b=*/false, b,
      /*c_mod=*/(short)0, c, /*reuse_a=*/false, /*reuse_b=*/false);
}

__device__ __forceinline__ float sigf(float x) { return 1.0f / (1.0f + __expf(-x)); }
__device__ __forceinline__ float tanhx(float x) {
  float e = __expf(2.0f * x);
  return (e - 1.0f) / (e + 1.0f);
}
__device__ __forceinline__ uint16_t f2bf(float f) {
  union { float f; uint32_t u; } v; v.f = f;
  uint32_t u = v.u + 0x7fffu + ((v.u >> 16) & 1u);  // RNE
  return (uint16_t)(u >> 16);
}

// u16 index of element (row m in tile, k kw in tile) inside packed WMMA-A buffer.
// A 16-bit 16x32 layout: lanes 0-15 hold K {0..7,16..23}, lanes 16-31 hold K {8..15,24..31}.
__device__ __forceinline__ unsigned a_idx(int mt, int kt, int m, int kw, int KT) {
  int e = kw & 1;
  int kk = kw & 15;
  int lane = m + ((kk & 8) ? 16 : 0);
  int j = ((kw >> 4) << 2) + ((kk & 7) >> 1);
  return ((unsigned)(mt * KT + kt) << 9) + (unsigned)(lane * 16 + j * 2 + e);
}

// ---------------- device-wide barrier (L2 atomics, sense-reversing) ----------------
__device__ void grid_barrier(unsigned* bar, unsigned nblocks) {
  __syncthreads();
  if (threadIdx.x == 0) {
    __threadfence();
    unsigned gen  = __hip_atomic_load(&bar[1], __ATOMIC_RELAXED, __HIP_MEMORY_SCOPE_AGENT);
    unsigned prev = __hip_atomic_fetch_add(&bar[0], 1u, __ATOMIC_ACQ_REL, __HIP_MEMORY_SCOPE_AGENT);
    if (prev == nblocks - 1) {
      __hip_atomic_store(&bar[0], 0u, __ATOMIC_RELAXED, __HIP_MEMORY_SCOPE_AGENT);
      __hip_atomic_fetch_add(&bar[1], 1u, __ATOMIC_ACQ_REL, __HIP_MEMORY_SCOPE_AGENT);
    } else {
      while (__hip_atomic_load(&bar[1], __ATOMIC_ACQUIRE, __HIP_MEMORY_SCOPE_AGENT) == gen)
        __builtin_amdgcn_s_sleep(2);
    }
    __threadfence();
  }
  __syncthreads();
}

// ---------------- pack x_t (f32 -> bf16, WMMA-A tile layout) into k-tiles 0..15 ----
__device__ __forceinline__ void pack_x(const float* __restrict__ x,
                                       uint16_t* __restrict__ A0dst, int t, int gt) {
  for (int i = gt; i < 4 * 16 * 512; i += cfg::NBLK * cfg::TPB) {
    int tile = i >> 9, r = i & 511;
    int mt = tile >> 4, kt = tile & 15;
    int ln = r >> 4, jj = (r >> 1) & 7, e = r & 1;
    int m = ln & 15;
    int kb = (ln >> 4) << 3;
    int K = kb + ((jj < 4) ? (jj * 2 + e) : (16 + ((jj - 4) << 1) + e));
    int brow = mt * 16 + m;
    int kcol = (kt << 5) + K;
    float v = GCAST(const float, x)[((size_t)brow * cfg::T + t) * cfg::DIN + kcol];
    GCAST(uint16_t, A0dst)[((size_t)(mt * cfg::KT0 + kt) << 9) + r] = f2bf(v);
  }
}

// ---------------- one LSTM layer step (GEMM + gates), K split over wave pairs -------
// KT etc. are compile-time so the k-loop unrolls and load offsets fold into the
// 24-bit immediate field of global_load_b128. B tiles are addressed from two
// anchors (gate0, gate2) so every gate offset fits the signed 24-bit immediate.
template <int KT, int HA_KT, int HA_KOFF, bool HASB, int HB_KT, int HB_KOFF>
__device__ __forceinline__ void lstm_layer_phase(
    const uint16_t* __restrict__ Abuf, const uint16_t* __restrict__ Wp,
    const float* __restrict__ bias, float scale, float* __restrict__ cbuf,
    uint16_t* __restrict__ hA, uint16_t* __restrict__ hB, float* red) {
  const int tid = threadIdx.x;
  const int w = tid >> 5, lane = tid & 31;
  const int gw = blockIdx.x * 8 + w;      // 0..511 global wave
  const int group = gw >> 1, half = gw & 1;
  const int mt = group >> 6, jt = group & 63;  // 4 m-tiles x 64 hidden-col tiles
  constexpr int KH = KT / 2;               // per-wave K share (wave pairs split K)
  constexpr int GSE = 64 * KT * 32;        // v16bf elems per gate block (<= 4MB)

  // per-lane base pointers into packed tile streams (stride 1KB per k-tile)
  const size_t laneoff = (size_t)lane * 16;
  auto* Av  = GCAST(const v16bf, Abuf + (((size_t)(mt * KT) + half * KH) << 9) + laneoff);
  auto* Blo = GCAST(const v16bf, Wp + (((size_t)((0 * 64 + jt) * KT) + half * KH) << 9) + laneoff);
  auto* Bhi = GCAST(const v16bf, Wp + (((size_t)((2 * 64 + jt) * KT) + half * KH) << 9) + laneoff);

  v8f aF = {}, aI = {}, aO = {}, aG = {};
#pragma unroll 4
  for (int kt = 0; kt < KH; ++kt) {
    v16bf a = Av[kt * 32];
    aF = wmma_bf16(a, Blo[kt * 32], aF);
    aI = wmma_bf16(a, Blo[kt * 32 + GSE], aI);
    aO = wmma_bf16(a, Bhi[kt * 32], aO);
    aG = wmma_bf16(a, Bhi[kt * 32 + GSE], aG);
  }

  const int p = w >> 1;
  float* rp = red + (size_t)p * 4 * 8 * 32;
  if (half) {
#pragma unroll
    for (int j = 0; j < 8; ++j) {
      rp[(0 * 8 + j) * 32 + lane] = aF[j];
      rp[(1 * 8 + j) * 32 + lane] = aI[j];
      rp[(2 * 8 + j) * 32 + lane] = aO[j];
      rp[(3 * 8 + j) * 32 + lane] = aG[j];
    }
  }
  __syncthreads();
  if (!half) {
    const int N = lane & 15;
    const int hcol = jt * 16 + N;
#pragma unroll
    for (int j = 0; j < 8; ++j) {
      float zf = (aF[j] + rp[(0 * 8 + j) * 32 + lane] + GCAST(const float, bias)[hcol]) * scale;
      float zi = (aI[j] + rp[(1 * 8 + j) * 32 + lane] + GCAST(const float, bias)[1024 + hcol]) * scale;
      float zo = (aO[j] + rp[(2 * 8 + j) * 32 + lane] + GCAST(const float, bias)[2048 + hcol]) * scale;
      float zg = (aG[j] + rp[(3 * 8 + j) * 32 + lane] + GCAST(const float, bias)[3072 + hcol]) * scale;
      float fg = sigf(zf), ig = sigf(zi), og = sigf(zo), gg = tanhx(zg);
      int mrow = ((lane >> 4) << 3) + j;           // D layout: M = (lane>=16?8:0)+j
      int brow = mt * 16 + mrow;
      float cold = GCAST(float, cbuf)[brow * 1024 + hcol];
      float cn = fg * cold + ig * gg;
      GCAST(float, cbuf)[brow * 1024 + hcol] = cn;
      uint16_t hb = f2bf(og * tanhx(cn));
      GCAST(uint16_t, hA)[a_idx(mt, HA_KOFF + (hcol >> 5), mrow, hcol & 31, HA_KT)] = hb;
      if constexpr (HASB)
        GCAST(uint16_t, hB)[a_idx(mt, HB_KOFF + (hcol >> 5), mrow, hcol & 31, HB_KT)] = hb;
    }
  }
}

// ---------------- persistent LSTM kernel ----------------
__global__ __launch_bounds__(cfg::TPB) void lstm_persist(
    const float* __restrict__ x, const float* __restrict__ b0,
    const float* __restrict__ b1, const float* __restrict__ bfc,
    const uint16_t* __restrict__ W0p, const uint16_t* __restrict__ W1p,
    const uint16_t* __restrict__ Wfcp,
    uint16_t* __restrict__ A0, uint16_t* __restrict__ A1,
    float* __restrict__ c0, float* __restrict__ c1,
    unsigned* __restrict__ bar, float* __restrict__ out) {
  __shared__ float red[4 * 4 * 8 * 32];  // 4 wave-pairs x 4 gates x 8 x 32
  const int tid = threadIdx.x;
  const int gt = blockIdx.x * cfg::TPB + tid;

  // prologue: pack x_0 into A0 parity 0
  pack_x(x, A0, 0, gt);
  grid_barrier(bar, cfg::NBLK);

  for (int t = 0; t < cfg::T; ++t) {
    const int par = t & 1;
    uint16_t* A0r = A0 + (size_t)par * cfg::A0E;        // layer0 A operand (x_t | h0(t-1))
    uint16_t* A0w = A0 + (size_t)(par ^ 1) * cfg::A0E;  // h0(t) -> layer0 at t+1
    uint16_t* A1r = A1 + (size_t)par * cfg::A1E;        // layer1 A operand (h0(t) | h1(t-1))
    uint16_t* A1w = A1 + (size_t)(par ^ 1) * cfg::A1E;  // h1(t) -> layer1 at t+1

    // phase 1: layer 0
    lstm_layer_phase<cfg::KT0, cfg::KT1, 0, true, cfg::KT0, 16>(
        A0r, W0p, b0, cfg::SCALE0, c0, A1r /*h0 -> layer1 input*/,
        A0w /*h0 -> layer0 next step*/, red);
    grid_barrier(bar, cfg::NBLK);

    // phase 2: layer 1, overlapped with packing x_{t+1} into A0[par^1] x-region
    lstm_layer_phase<cfg::KT1, cfg::KT1, 32, false, 1, 0>(
        A1r, W1p, b1, cfg::SCALE1, c1, A1w /*h1 -> layer1 next step*/,
        nullptr, red);
    if (t + 1 < cfg::T) pack_x(x, A0w, t + 1, gt);
    grid_barrier(bar, cfg::NBLK);
  }

  // final FC: out = h1(T-1) @ Wfc + bfc ; h1(T-1) sits in A1 parity 0, k-tiles 32..63
  if (blockIdx.x < 32) {
    const int w = tid >> 5, lane = tid & 31;
    const int gw = blockIdx.x * 8 + w;     // 0..255
    const int group = gw >> 1, half = gw & 1;
    const int mt = group >> 5, nt = group & 31;
    constexpr int KH = cfg::KTFC / 2;      // 16
    const size_t laneoff = (size_t)lane * 16;
    auto* Av = GCAST(const v16bf,
                     A1 + (((size_t)(mt * cfg::KT1 + 32) + half * KH) << 9) + laneoff);
    auto* Bv = GCAST(const v16bf,
                     Wfcp + (((size_t)(nt * cfg::KTFC) + half * KH) << 9) + laneoff);
    v8f acc = {};
#pragma unroll 4
    for (int kt = 0; kt < KH; ++kt)
      acc = wmma_bf16(Av[kt * 32], Bv[kt * 32], acc);
    const int p = w >> 1;
    float* rp = red + (size_t)p * 4 * 8 * 32;
    if (half) {
#pragma unroll
      for (int j = 0; j < 8; ++j) rp[j * 32 + lane] = acc[j];
    }
    __syncthreads();
    if (!half) {
      const int N = lane & 15;
#pragma unroll
      for (int j = 0; j < 8; ++j) {
        int mrow = ((lane >> 4) << 3) + j;
        int brow = mt * 16 + mrow;
        int col = nt * 16 + N;
        GCAST(float, out)[brow * cfg::DOUT + col] =
            acc[j] + rp[j * 32 + lane] + GCAST(const float, bfc)[col];
      }
    }
  }
}

// ---------------- weight pack: f32 row-major [K,N] -> bf16 WMMA-B tiles ----------------
// B 16-bit 32x16 layout: lane = n + ((kw&8) ? 16 : 0); vgpr j = ((kw>>4)<<2) + ((kw&7)>>1); half e = kw&1.
__global__ void pack_weight(const float* __restrict__ W, uint16_t* __restrict__ outp,
                            int K, int N) {
  const int KT = K >> 5;
  const long total = (long)K * N;
  for (long i = (long)blockIdx.x * blockDim.x + threadIdx.x; i < total;
       i += (long)gridDim.x * blockDim.x) {
    int k = (int)(i / N), n = (int)(i % N);
    int kt = k >> 5, nt = n >> 4, kw = k & 31, nw = n & 15;
    int lane = nw + ((kw & 8) ? 16 : 0);
    int j = ((kw >> 4) << 2) + ((kw & 7) >> 1), e = kw & 1;
    GCAST(uint16_t, outp)[((size_t)(nt * KT + kt) << 9) + lane * 16 + j * 2 + e] =
        f2bf(GCAST(const float, W)[i]);
  }
}

__global__ void zero_u32(uint32_t* __restrict__ p, int n) {
  for (int i = blockIdx.x * blockDim.x + threadIdx.x; i < n;
       i += gridDim.x * blockDim.x)
    GCAST(uint32_t, p)[i] = 0u;
}

extern "C" void kernel_launch(void* const* d_in, const int* in_sizes, int n_in,
                              void* d_out, int out_size, void* d_ws, size_t ws_size,
                              hipStream_t stream) {
  (void)in_sizes; (void)n_in; (void)out_size; (void)ws_size;
  const float* x   = (const float*)d_in[0];
  const float* W0  = (const float*)d_in[1];
  const float* b0  = (const float*)d_in[2];
  const float* W1  = (const float*)d_in[3];
  const float* b1  = (const float*)d_in[4];
  const float* Wfc = (const float*)d_in[5];
  const float* bfc = (const float*)d_in[6];

  char* ws = (char*)d_ws;
  size_t off = 0;
  uint16_t* W0p  = (uint16_t*)(ws + off); off += (size_t)cfg::KT0  * cfg::NT01 * 1024;
  uint16_t* W1p  = (uint16_t*)(ws + off); off += (size_t)cfg::KT1  * cfg::NT01 * 1024;
  uint16_t* Wfcp = (uint16_t*)(ws + off); off += (size_t)cfg::KTFC * cfg::NTFC * 1024;
  char* state = ws + off;
  uint16_t* A0 = (uint16_t*)(ws + off); off += 2 * cfg::A0E * sizeof(uint16_t);
  uint16_t* A1 = (uint16_t*)(ws + off); off += 2 * cfg::A1E * sizeof(uint16_t);
  float* c0 = (float*)(ws + off); off += (size_t)cfg::B * cfg::H * sizeof(float);
  float* c1 = (float*)(ws + off); off += (size_t)cfg::B * cfg::H * sizeof(float);
  unsigned* bar = (unsigned*)(ws + off); off += 256;
  const size_t state_bytes = (size_t)((ws + off) - state);

  zero_u32<<<512, 256, 0, stream>>>((uint32_t*)state, (int)(state_bytes / 4));
  pack_weight<<<2048, 256, 0, stream>>>(W0, W0p, 1536, 4096);
  pack_weight<<<2048, 256, 0, stream>>>(W1, W1p, 2048, 4096);
  pack_weight<<<512, 256, 0, stream>>>(Wfc, Wfcp, 1024, 512);
  lstm_persist<<<cfg::NBLK, cfg::TPB, 0, stream>>>(
      x, b0, b1, bfc, W0p, W1p, Wfcp, A0, A1, c0, c1, bar, (float*)d_out);
}
